// GridSampleCrossBEVAttention_3753801416838
// MI455X (gfx1250) — matrix operand order repaired and checked by reading, not compiled, not run on hardware
//
#include <hip/hip_runtime.h>

typedef __attribute__((ext_vector_type(16))) _Float16 v16h;
typedef __attribute__((ext_vector_type(8)))  float    v8f;

#define H_IMG  200
#define W_IMG  200
#define CIN    64
#define DM     256
#define NQ     2048
#define NP     8
#define BSZ    8

union FragU { uint4 u[2]; v16h h; _Float16 f[16]; };

// Assemble a 16-f16 WMMA fragment from two contiguous 16B runs.
static __device__ inline v16h load_frag_pair(const _Float16* p0, const _Float16* p1) {
    FragU u;
    u.u[0] = *reinterpret_cast<const uint4*>(p0);
    u.u[1] = *reinterpret_cast<const uint4*>(p1);
    return u.h;
}

// Async global->LDS copy of 16 bytes; IMM offset applies to BOTH addresses (ISA 15.18.3).
#define ASYNC_LDS_CP16(ldsoff, gptr, IMM)                                     \
    asm volatile("global_load_async_to_lds_b128 %0, %1, off offset:" IMM      \
                 :: "v"(ldsoff), "v"(gptr) : "memory")

// ---------------- prep: bev NCHW f32 -> NHWC f16 ----------------
__global__ void prep_bev_kernel(const float* __restrict__ in, _Float16* __restrict__ out, int n) {
    int i = blockIdx.x * 256 + threadIdx.x;
    if (i >= n) return;
    int c = i & (CIN - 1);
    int x = (i >> 6) % W_IMG;
    int t = (i >> 6) / W_IMG;
    int y = t % H_IMG;
    int b = t / H_IMG;
    out[i] = (_Float16)in[(((b * CIN + c) * H_IMG + y) * W_IMG) + x];
}

// ---------------- prep: conv_w (O,I,3,3) f32 -> WMMA B-fragment order f16 ----------------
// pack index: (((kk*16 + nsub)*32 + lane)*16 + e), kk = tap*2 + kc (K32 chunks, im2col K=576)
// B layout (32x16 KxN): N = nsub*16 + (lane&15); K-in-chunk = 16*(lane>>4) + e
__global__ void prep_convw_kernel(const float* __restrict__ w, _Float16* __restrict__ pack) {
    int i = blockIdx.x * 256 + threadIdx.x;        // 18*16*32*16 = 147456
    if (i >= 18 * 16 * 32 * 16) return;
    int e    = i & 15;
    int lane = (i >> 4) & 31;
    int nsub = (i >> 9) & 15;
    int kk   = i >> 13;                            // 0..17
    int tap  = kk >> 1, kc = kk & 1;
    int ci   = kc * 32 + ((lane >> 4) * 16) + e;   // 0..63
    int ky   = tap / 3, kx = tap % 3;
    int co   = nsub * 16 + (lane & 15);
    pack[i] = (_Float16)w[(((co * CIN + ci) * 3 + ky) * 3) + kx];
}

// ---------------- prep: out_w (D,C) f32 -> WMMA B-fragment order f16 (B[K=c][N=d]=out_w[d][c]) ----
__global__ void prep_outw_kernel(const float* __restrict__ w, _Float16* __restrict__ pack) {
    int i = blockIdx.x * 256 + threadIdx.x;        // 8*16*32*16 = 65536
    if (i >= 8 * 16 * 32 * 16) return;
    int e    = i & 15;
    int lane = (i >> 4) & 31;
    int nsub = (i >> 9) & 15;
    int kc   = i >> 13;                            // 0..7
    int c    = kc * 32 + ((lane >> 4) * 16) + e;
    int d    = nsub * 16 + (lane & 15);
    pack[i] = (_Float16)w[d * DM + c];
}

// ---------------- attention weights: softmax(queries @ attn_w^T + attn_b) ----------------
__global__ void attw_kernel(const float* __restrict__ queries, const float* __restrict__ attn_w,
                            const float* __restrict__ attn_b, float* __restrict__ attw) {
    int bq = blockIdx.x * 256 + threadIdx.x;
    if (bq >= BSZ * NQ) return;
    float l[NP];
#pragma unroll
    for (int p = 0; p < NP; ++p) l[p] = attn_b[p];
    const float* qrow = &queries[bq * DM];
    for (int c = 0; c < DM; ++c) {
        float qv = qrow[c];
#pragma unroll
        for (int p = 0; p < NP; ++p) l[p] += qv * attn_w[p * DM + c];
    }
    float mx = l[0];
#pragma unroll
    for (int p = 1; p < NP; ++p) mx = fmaxf(mx, l[p]);
    float s = 0.f, ex[NP];
#pragma unroll
    for (int p = 0; p < NP; ++p) { ex[p] = __expf(l[p] - mx); s += ex[p]; }
    float inv = 1.0f / s;
#pragma unroll
    for (int p = 0; p < NP; ++p) attw[bq * NP + p] = ex[p] * inv;
}

// ---------------- conv3x3 + bias + relu, implicit GEMM with WMMA ----------------
// grid = (4 x-tiles, 200 rows, 8 batches); block = 256 (8 waves)
// Wave tiling: each wave computes 64 px (4 M-subtiles) x 32 ch (2 N-subtiles)
//   -> per K-chunk only 2 B frags (loaded by exactly ONE wave per WG) + 4 A frags (LDS broadcast)
// LDS (32KB) reused: phase 1 = input halo (3 x 66 x 64 f16), phase 2 = output tile (64 x 256 f16)
__global__ __launch_bounds__(256) void conv_kernel(const _Float16* __restrict__ bev,
                                                   const _Float16* __restrict__ wpack,
                                                   const float* __restrict__ conv_b,
                                                   _Float16* __restrict__ value) {
    __shared__ __align__(16) _Float16 smem[64 * DM];   // 32 KB
    const int x0  = blockIdx.x * 64;
    const int y   = blockIdx.y;
    const int b   = blockIdx.z;
    const int tid = threadIdx.x;

    // ---- stage input halo (zero padded) via async global->LDS (ASYNCcnt path) ----
    for (int idx = tid; idx < 3 * 66; idx += 256) {
        const int r  = idx / 66;
        const int xc = idx % 66;
        const int yi = y + r - 1;
        const int xi = x0 + xc - 1;
        if (yi >= 0 && yi < H_IMG && xi >= 0 && xi < W_IMG) {
            const unsigned lds_off = (unsigned)(uintptr_t)(const void*)&smem[idx * CIN];
            const void* gp = (const void*)&bev[((b * H_IMG + yi) * W_IMG + xi) * CIN];
            ASYNC_LDS_CP16(lds_off, gp, "0");
            ASYNC_LDS_CP16(lds_off, gp, "16");
            ASYNC_LDS_CP16(lds_off, gp, "32");
            ASYNC_LDS_CP16(lds_off, gp, "48");
            ASYNC_LDS_CP16(lds_off, gp, "64");
            ASYNC_LDS_CP16(lds_off, gp, "80");
            ASYNC_LDS_CP16(lds_off, gp, "96");
            ASYNC_LDS_CP16(lds_off, gp, "112");
        } else {
            uint4 z = make_uint4(0u, 0u, 0u, 0u);
            uint4* dst = reinterpret_cast<uint4*>(&smem[idx * CIN]);
#pragma unroll
            for (int k = 0; k < 8; ++k) dst[k] = z;
        }
    }
    asm volatile("s_wait_asynccnt 0x0" ::: "memory");
    __syncthreads();

    const int lane = tid & 31;
    const int wid  = tid >> 5;      // 8 waves, wave owns N-subtile pair (wid*2, wid*2+1)
    const int m = lane & 15;
    const int h = lane >> 4;

    v8f acc[4][2];
#pragma unroll
    for (int mi = 0; mi < 4; ++mi) {
        acc[mi][0] = {};
        acc[mi][1] = {};
    }

    // B-fragment offset for this wave (int offsets -> SADDR+voffset addressing)
    int bo = ((wid * 2) * 32 + lane) * 16;          // +512 for nsub+1, +8192 per K-chunk

    for (int kk = 0; kk < 18; ++kk) {
        const int tap = kk >> 1, kc = kk & 1;
        const int ky = tap / 3, kx = tap - ky * 3;
        // 2 B fragments (4 x b128, each fragment unique to this wave within the WG)
        v16h bf0 = load_frag_pair(wpack + bo,       wpack + bo + 8);
        v16h bf1 = load_frag_pair(wpack + bo + 512, wpack + bo + 520);
        // 4 A fragments: 4 pixel subtiles x 32 ci from LDS (two 16B runs each)
        const int abase = (ky * 66 + m + kx) * CIN + kc * 32 + 16 * h;
        v16h af[4];
#pragma unroll
        for (int mi = 0; mi < 4; ++mi) {
            const _Float16* ab = &smem[abase + mi * 16 * CIN];
            af[mi] = load_frag_pair(ab, ab + 8);
        }
#pragma unroll
        for (int mi = 0; mi < 4; ++mi) {
            acc[mi][0] = __builtin_amdgcn_wmma_f32_16x16x32_f16(
                false, af[mi], false, bf0, (short)0, acc[mi][0], false, false);
            acc[mi][1] = __builtin_amdgcn_wmma_f32_16x16x32_f16(
                false, af[mi], false, bf1, (short)0, acc[mi][1], false, false);
        }
        bo += 8192;
    }

    // ---- epilogue: bias + relu -> LDS tile, then coalesced b128 stores (NHWC f16) ----
    __syncthreads();   // input staging dead; reuse smem as 64x256 output tile
#pragma unroll
    for (int j = 0; j < 2; ++j) {
        const int co = (wid * 2 + j) * 16 + (lane & 15);
        const float bias = conv_b[co];
#pragma unroll
        for (int mi = 0; mi < 4; ++mi) {
#pragma unroll
            for (int r = 0; r < 8; ++r) {
                float v = acc[mi][j][r] + bias;
                v = v > 0.0f ? v : 0.0f;
                smem[(mi * 16 + r + 8 * h) * DM + co] = (_Float16)v;
            }
        }
    }
    __syncthreads();
    const int vbase = ((b * H_IMG + y) * W_IMG + x0) * DM;
    for (int task = tid; task < 64 * 16; task += 256) {
        const int px  = task >> 4;
        const int ch0 = (task & 15) * 16;
        if (x0 + px < W_IMG) {
            const uint4* s = reinterpret_cast<const uint4*>(&smem[px * DM + ch0]);
            uint4* d = reinterpret_cast<uint4*>(value + (vbase + px * DM + ch0));
            d[0] = s[0];
            d[1] = s[1];
        }
    }
}

// ---------------- fused bilinear-sample + point-mix + output projection + residual ----------------
// block = 256 threads, 16 queries per block; projection is WMMA K=256
__global__ __launch_bounds__(256) void sample_proj_kernel(const float* __restrict__ traj,
                                                          const float* __restrict__ attw,
                                                          const _Float16* __restrict__ value,
                                                          const _Float16* __restrict__ owpack,
                                                          const float* __restrict__ out_b,
                                                          const float* __restrict__ queries,
                                                          float* __restrict__ out) {
    __shared__ __align__(16) _Float16 ctx[16 * DM];   // 16 queries x 256 ch (f16 A-staging)
    const int tid = threadIdx.x;
    const int bq0 = blockIdx.x * 16;
    const int b   = bq0 >> 11;                        // / 2048 (NQ)
    const int ql  = tid >> 4;                         // 0..15 query in tile
    const int c0  = (tid & 15) * 16;                  // 16-channel chunk
    const int bq  = bq0 + ql;

    float acc16[16];
#pragma unroll
    for (int k = 0; k < 16; ++k) acc16[k] = 0.f;

    for (int p = 0; p < NP; ++p) {
        const float tx = traj[(bq * NP + p) * 2 + 0];
        const float ty = traj[(bq * NP + p) * 2 + 1];
        const float g0 = ty * (1.0f / 30.0f);   // grid x  (reversed last dim)
        const float g1 = tx * (1.0f / 30.0f);   // grid y
        const float gx = (g0 + 1.0f) * 0.5f * (float)(W_IMG - 1);
        const float gy = (g1 + 1.0f) * 0.5f * (float)(H_IMG - 1);
        int x0i = (int)gx; x0i = x0i < 0 ? 0 : (x0i > W_IMG - 2 ? W_IMG - 2 : x0i);
        int y0i = (int)gy; y0i = y0i < 0 ? 0 : (y0i > H_IMG - 2 ? H_IMG - 2 : y0i);
        float wx = fminf(fmaxf(gx - (float)x0i, 0.f), 1.f);
        float wy = fminf(fmaxf(gy - (float)y0i, 0.f), 1.f);
        const bool inb = (g0 >= -1.f) && (g0 <= 1.f) && (g1 >= -1.f) && (g1 <= 1.f);
        const float aw  = attw[bq * NP + p] * (inb ? 1.f : 0.f);
        const float w00 = aw * (1.f - wx) * (1.f - wy);
        const float w01 = aw * wx * (1.f - wy);
        const float w10 = aw * (1.f - wx) * wy;
        const float w11 = aw * wx * wy;

        const int vo = ((b * H_IMG + y0i) * W_IMG + x0i) * DM + c0;
        const _Float16* v00p = value + vo;
        const _Float16* v01p = v00p + DM;
        const _Float16* v10p = v00p + W_IMG * DM;
        const _Float16* v11p = v10p + DM;
        FragU u00, u01, u10, u11;
        u00.u[0] = ((const uint4*)v00p)[0]; u00.u[1] = ((const uint4*)v00p)[1];
        u01.u[0] = ((const uint4*)v01p)[0]; u01.u[1] = ((const uint4*)v01p)[1];
        u10.u[0] = ((const uint4*)v10p)[0]; u10.u[1] = ((const uint4*)v10p)[1];
        u11.u[0] = ((const uint4*)v11p)[0]; u11.u[1] = ((const uint4*)v11p)[1];
#pragma unroll
        for (int k = 0; k < 16; ++k) {
            acc16[k] += w00 * (float)u00.f[k] + w01 * (float)u01.f[k]
                      + w10 * (float)u10.f[k] + w11 * (float)u11.f[k];
        }
    }
#pragma unroll
    for (int k = 0; k < 16; ++k) ctx[ql * DM + c0 + k] = (_Float16)acc16[k];
    __syncthreads();

    // projection: out[q][d] = sum_c ctx[q][c] * out_w[d][c]  (WMMA, K = 256)
    const int lane = tid & 31;
    const int wid  = tid >> 5;              // 8 waves, each covers 32 outputs
    const int h    = lane >> 4;
    v8f acc0 = {}, acc1 = {};
    int b0o = ((wid * 2) * 32 + lane) * 16;           // +512 for ns1, +8192 per kc
    for (int kc = 0; kc < 8; ++kc) {
        v16h bf0 = load_frag_pair(owpack + b0o,       owpack + b0o + 8);
        v16h bf1 = load_frag_pair(owpack + b0o + 512, owpack + b0o + 520);
        const _Float16* ab = &ctx[(lane & 15) * DM + kc * 32 + 16 * h];
        v16h afrag = load_frag_pair(ab, ab + 8);
        acc0 = __builtin_amdgcn_wmma_f32_16x16x32_f16(false, afrag, false, bf0, (short)0, acc0, false, false);
        acc1 = __builtin_amdgcn_wmma_f32_16x16x32_f16(false, afrag, false, bf1, (short)0, acc1, false, false);
        b0o += 8192;
    }

    // epilogue: + out_b + residual, f32 store
    const int d0 = (wid * 2) * 16 + (lane & 15);
    const int d1 = (wid * 2 + 1) * 16 + (lane & 15);
#pragma unroll
    for (int r = 0; r < 8; ++r) {
        const int qm    = r + 8 * h;
        const int obase = (bq0 + qm) * DM;
        out[obase + d0] = acc0[r] + out_b[d0] + queries[obase + d0];
        out[obase + d1] = acc1[r] + out_b[d1] + queries[obase + d1];
    }
}

extern "C" void kernel_launch(void* const* d_in, const int* in_sizes, int n_in,
                              void* d_out, int out_size, void* d_ws, size_t ws_size,
                              hipStream_t stream) {
    const float* queries = (const float*)d_in[0];   // (8,2048,256)
    const float* traj    = (const float*)d_in[1];   // (8,2048,8,2)
    const float* bev     = (const float*)d_in[2];   // (8,64,200,200)
    const float* attn_w  = (const float*)d_in[3];   // (8,256)
    const float* attn_b  = (const float*)d_in[4];   // (8,)
    const float* out_w   = (const float*)d_in[5];   // (256,256)
    const float* out_b   = (const float*)d_in[6];   // (256,)
    const float* conv_w  = (const float*)d_in[7];   // (256,64,3,3)
    const float* conv_b  = (const float*)d_in[8];   // (256,)
    float* out = (float*)d_out;                     // (8,2048,256)

    // workspace carve-out (256B aligned)
    char* ws = (char*)d_ws;
    size_t off = 0;
    auto carve = [&](size_t bytes) {
        void* p = ws + off;
        off = (off + bytes + 255) & ~(size_t)255;
        return p;
    };
    _Float16* bev16  = (_Float16*)carve((size_t)BSZ * H_IMG * W_IMG * CIN * 2);   // 41 MB
    _Float16* val16  = (_Float16*)carve((size_t)BSZ * H_IMG * W_IMG * DM * 2);    // 164 MB (fits L2)
    _Float16* wpack  = (_Float16*)carve((size_t)18 * 16 * 32 * 16 * 2);           // 288 KB
    _Float16* owpack = (_Float16*)carve((size_t)8 * 16 * 32 * 16 * 2);            // 128 KB
    float*    attwb  = (float*)carve((size_t)BSZ * NQ * NP * 4);                  // 512 KB
    (void)ws_size; (void)in_sizes; (void)n_in; (void)out_size;

    const int nbev = BSZ * H_IMG * W_IMG * CIN;
    prep_bev_kernel<<<(nbev + 255) / 256, 256, 0, stream>>>(bev, bev16, nbev);
    prep_convw_kernel<<<(18 * 16 * 32 * 16 + 255) / 256, 256, 0, stream>>>(conv_w, wpack);
    prep_outw_kernel<<<(8 * 16 * 32 * 16 + 255) / 256, 256, 0, stream>>>(out_w, owpack);
    attw_kernel<<<(BSZ * NQ + 255) / 256, 256, 0, stream>>>(queries, attn_w, attn_b, attwb);

    dim3 cgrid(4, H_IMG, BSZ);   // 64-pixel x-tiles, rows, batches
    conv_kernel<<<cgrid, 256, 0, stream>>>(bev16, wpack, conv_b, val16);

    sample_proj_kernel<<<(BSZ * NQ) / 16, 256, 0, stream>>>(traj, attwb, val16, owpack,
                                                            out_b, queries, out);
}
